// AttentionBlock_27625229648173
// MI455X (gfx1250) — matrix operand order repaired and checked by reading, not compiled, not run on hardware
//
#include <hip/hip_runtime.h>
#include <hip/hip_bf16.h>
#include <math.h>

typedef __attribute__((ext_vector_type(16))) __bf16       v16bf;
typedef __attribute__((ext_vector_type(8)))  float        v8f;
typedef __attribute__((ext_vector_type(4)))  unsigned int u32x4;

union FragBF { u32x4 q[2]; v16bf bf; };        // 32 bytes: one WMMA bf16 operand
union Pack8  { unsigned short s[8]; u32x4 q; };

#define B_BATCH 64
#define NPIX    196      // 14*14
#define CIN     2048
#define NH1     64
#define TILE_M  16
#define NTILES  13       // ceil(196/16)
#define KBLKS   (CIN/32) // 64 K-steps of 32

__device__ __forceinline__ unsigned short f2bf(float f) {
    unsigned int u = __builtin_bit_cast(unsigned int, f);
    u += 0x7FFFu + ((u >> 16) & 1u);           // round-to-nearest-even
    return (unsigned short)(u >> 16);
}

// Pack W1 [2048,64] fp32 -> bf16 in WMMA B-fragment lane order.
// ushort index t = ((kb*4 + ntile)*32 + lane)*16 + j
//   lanes 0-15 hold K = kb*32 + j        (col n = ntile*16 + lane)
//   lanes 16-31 hold K = kb*32 + 16 + j  (col n = ntile*16 + lane-16)
__global__ void pack_w1(const float* __restrict__ W1, unsigned short* __restrict__ dst) {
    int t = blockIdx.x * blockDim.x + threadIdx.x;
    if (t >= CIN * NH1) return;
    int j  = t & 15;
    int l  = (t >> 4) & 31;
    int nt = (t >> 9) & 3;
    int kb = t >> 11;
    int K  = kb * 32 + ((l >> 4) << 4) + j;
    int n  = nt * 16 + (l & 15);
    dst[t] = f2bf(W1[K * NH1 + n]);
}

__global__ void zero_acc(float* __restrict__ p, int n) {
    int i = blockIdx.x * blockDim.x + threadIdx.x;
    if (i < n) p[i] = 0.0f;
}

__global__ __launch_bounds__(128)
void attn_main(const float* __restrict__ x,
               const float* __restrict__ b1,
               const float* __restrict__ W2, const float* __restrict__ b2,
               const float* __restrict__ W3, const float* __restrict__ b3,
               const float* __restrict__ W4, const float* __restrict__ b4,
               const unsigned short* __restrict__ w1p,
               float* __restrict__ feat_acc, float* __restrict__ mask_acc) {
    // 64KB LDS: x-tile bf16, chunk layout [CIN/8][16 rows][8 halfs] -> conflict-free b128 A loads
    __shared__ unsigned char smem[65536];
    u32x4* xq = (u32x4*)smem;                 // 16B chunk (c,m) at index c*16+m

    const int tid  = threadIdx.x;
    const int b    = blockIdx.y;
    const int tile = blockIdx.x;
    const int gp0  = tile * TILE_M;
    int nvalid = NPIX - gp0; if (nvalid > TILE_M) nvalid = TILE_M;

    // ---- stage x tile (fp32 global -> bf16 LDS), zero-pad invalid rows ----
    for (int it = 0; it < 32; ++it) {
        int idx = it * 128 + tid;             // 4096 chunks of 8 floats
        int m = idx >> 8;
        int c = idx & 255;
        float v[8];
        if (m < nvalid) {
            const float4* s4 = (const float4*)(x + ((size_t)(b * NPIX + gp0 + m) * CIN + c * 8));
            float4 a0 = s4[0], a1 = s4[1];
            v[0]=a0.x; v[1]=a0.y; v[2]=a0.z; v[3]=a0.w;
            v[4]=a1.x; v[5]=a1.y; v[6]=a1.z; v[7]=a1.w;
        } else {
            #pragma unroll
            for (int q = 0; q < 8; ++q) v[q] = 0.0f;
        }
        Pack8 p;
        #pragma unroll
        for (int q = 0; q < 8; ++q) p.s[q] = f2bf(v[q]);
        xq[c * 16 + m] = p.q;
    }
    __syncthreads();

    // ---- GEMM1: each wave computes 16x16 of h1 = Xtile(16x2048) @ W1(:,16w:16w+16) ----
    const int w  = tid >> 5;                  // wave = N-tile
    const int l  = tid & 31;
    const int m  = l & 15;                    // A: lane -> row M ; C/D: lane -> col N
    const int hi = l >> 4;
    v8f acc = {0.f,0.f,0.f,0.f,0.f,0.f,0.f,0.f};
    const u32x4* bq = (const u32x4*)w1p;
    #pragma unroll 2
    for (int kb = 0; kb < KBLKS; ++kb) {
        FragBF A, Bf;
        int c0 = (kb << 2) + (hi << 1);       // chunk = k0/8 (+2 for hi half-wave)
        A.q[0] = xq[(c0    ) * 16 + m];       // K = k0 + 16*hi .. +7
        A.q[1] = xq[(c0 + 1) * 16 + m];       // K = k0+16*hi+8 .. +7
        int bidx = ((kb * 4 + w) * 32 + l) * 2;
        Bf.q[0] = bq[bidx];
        Bf.q[1] = bq[bidx + 1];
        acc = __builtin_amdgcn_wmma_f32_16x16x32_bf16(false, A.bf, false, Bf.bf,
                                                      (short)0, acc, false, false);
    }
    __syncthreads();                          // x-tile dead; reuse LDS

    float* h1  = (float*)smem;                // [16][64]
    float* a_s = (float*)(smem + TILE_M * NH1 * 4);
    {
        int n = w * 16 + m;                   // h1 channel
        float bias = b1[n];
        #pragma unroll
        for (int r = 0; r < 8; ++r) {         // C/D layout: vgpr r -> M = r (+8 hi lanes)
            float v = acc[r] + bias;
            h1[(r + hi * 8) * NH1 + n] = v > 0.f ? v : 0.f;
        }
    }
    __syncthreads();

    // ---- tiny per-pixel MLP (64->16->8->1, sigmoid): one thread per pixel ----
    if (tid < TILE_M) {
        int p = tid;
        float h2[16];
        #pragma unroll
        for (int j = 0; j < 16; ++j) {
            float s = b2[j];
            for (int i = 0; i < 64; ++i) s += h1[p * NH1 + i] * W2[i * 16 + j];
            h2[j] = s > 0.f ? s : 0.f;
        }
        float h3[8];
        #pragma unroll
        for (int j = 0; j < 8; ++j) {
            float s = b3[j];
            #pragma unroll
            for (int i = 0; i < 16; ++i) s += h2[i] * W3[i * 8 + j];
            h3[j] = s > 0.f ? s : 0.f;
        }
        float t = b4[0];
        #pragma unroll
        for (int i = 0; i < 8; ++i) t += h3[i] * W4[i];
        a_s[p] = 1.0f / (1.0f + __expf(-t));
    }
    __syncthreads();

    // ---- accumulate sum(a) and sum(a*x) per batch (x re-read fp32, L2-hot) ----
    if (tid == 0) {
        float s = 0.f;
        for (int p = 0; p < nvalid; ++p) s += a_s[p];
        atomicAdd(&mask_acc[b], s);
    }
    for (int c = tid; c < CIN; c += 128) {
        float f = 0.f;
        for (int p = 0; p < nvalid; ++p)
            f += a_s[p] * x[(size_t)(b * NPIX + gp0 + p) * CIN + c];
        atomicAdd(&feat_acc[(size_t)b * CIN + c], f);
    }
}

__global__ void finalize_div(const float* __restrict__ feat, const float* __restrict__ maskv,
                             float* __restrict__ out) {
    int i = blockIdx.x * blockDim.x + threadIdx.x;
    if (i < B_BATCH * CIN) out[i] = feat[i] / maskv[i >> 11];   // 1/196 cancels
}

extern "C" void kernel_launch(void* const* d_in, const int* in_sizes, int n_in,
                              void* d_out, int out_size, void* d_ws, size_t ws_size,
                              hipStream_t stream) {
    const float* x  = (const float*)d_in[0];
    const float* W1 = (const float*)d_in[1];
    const float* b1 = (const float*)d_in[2];
    const float* W2 = (const float*)d_in[3];
    const float* b2 = (const float*)d_in[4];
    const float* W3 = (const float*)d_in[5];
    const float* b3 = (const float*)d_in[6];
    const float* W4 = (const float*)d_in[7];
    const float* b4 = (const float*)d_in[8];

    unsigned short* w1p  = (unsigned short*)d_ws;              // 256 KB packed bf16 W1
    float*          feat = (float*)((char*)d_ws + 262144);     // 512 KB feature accumulators
    float*          mask = feat + B_BATCH * CIN;               // 256 B mask accumulators

    pack_w1<<<(CIN * NH1 + 255) / 256, 256, 0, stream>>>(W1, w1p);
    zero_acc<<<(B_BATCH * CIN + B_BATCH + 255) / 256, 256, 0, stream>>>(feat, B_BATCH * CIN + B_BATCH);
    attn_main<<<dim3(NTILES, B_BATCH), 128, 0, stream>>>(x, b1, W2, b2, W3, b3, W4, b4,
                                                         w1p, feat, mask);
    finalize_div<<<(B_BATCH * CIN + 255) / 256, 256, 0, stream>>>(feat, mask, (float*)d_out);
}
